// SparseDynamicConv3d_46342697124229
// MI455X (gfx1250) — matrix operand order repaired and checked by reading, not compiled, not run on hardware
//
#include <hip/hip_runtime.h>
#include <hip/hip_bf16.h>

typedef __attribute__((ext_vector_type(16))) __bf16 v16bf;
typedef __attribute__((ext_vector_type(8)))  float  v8f;
typedef __attribute__((ext_vector_type(4)))  int    v4i;

#define KVOL   27
#define MPTS   80000
#define RIN    48
#define ROUT   96
#define INC    64
#define OUTC   128
#define MTILE  128
#define KPAD   64        // rin padded to 64 for two K=32 WMMA steps
#define AS_STRIDE 72
#define BS_STRIDE 72     // padded LDS stride (ushorts) to avoid bank conflicts

#define AS1 __attribute__((address_space(1)))
#define AS3 __attribute__((address_space(3)))

#if defined(__has_builtin)
#if __has_builtin(__builtin_amdgcn_global_load_async_to_lds_b128)
#define HAVE_ASYNC_LDS 1
#endif
#endif

__device__ __forceinline__ unsigned short f32_to_bf16_rne(float f) {
    unsigned int x = __float_as_uint(f);
    unsigned int r = x + 0x7FFFu + ((x >> 16) & 1u);
    return (unsigned short)(r >> 16);
}

union BFrag { v16bf v; uint4 q[2]; unsigned int u[8]; };

__global__ void __launch_bounds__(256) zero_out_kernel(float4* out, int n4) {
    int i = blockIdx.x * blockDim.x + threadIdx.x;
    if (i < n4) out[i] = make_float4(0.f, 0.f, 0.f, 0.f);
}

// ---- Prep: features f32[N][48] -> bf16[N][64] (K zero-padded), 2 threads/row ----
__global__ void __launch_bounds__(256)
feat_bf16_kernel(const float* __restrict__ f, unsigned short* __restrict__ o, int nRows) {
    int tid  = blockIdx.x * 256 + threadIdx.x;
    int row  = tid >> 1;
    int part = tid & 1;
    if (row >= nRows) return;
    const float4* src = (const float4*)(f + (size_t)row * RIN + part * 24);
    unsigned short* dst = o + (size_t)row * KPAD + part * 24;
    #pragma unroll
    for (int j = 0; j < 6; ++j) {
        float4 v = src[j];
        dst[j * 4 + 0] = f32_to_bf16_rne(v.x);
        dst[j * 4 + 1] = f32_to_bf16_rne(v.y);
        dst[j * 4 + 2] = f32_to_bf16_rne(v.z);
        dst[j * 4 + 3] = f32_to_bf16_rne(v.w);
    }
    if (part) {
        unsigned short* pad = o + (size_t)row * KPAD + RIN;
        #pragma unroll
        for (int j = 0; j < KPAD - RIN; ++j) pad[j] = 0;
    }
}

// ---- Prep: weights f32[27][64][128] -> bf16[27][96][72] transposed [o][kk], padded ----
__global__ void __launch_bounds__(256)
weight_bf16_kernel(const float* __restrict__ kern, unsigned short* __restrict__ wB) {
    int idx = blockIdx.x * 256 + threadIdx.x;
    if (idx >= KVOL * ROUT * BS_STRIDE) return;
    int k   = idx / (ROUT * BS_STRIDE);
    int rem = idx % (ROUT * BS_STRIDE);
    int o   = rem / BS_STRIDE;
    int kk  = rem % BS_STRIDE;
    unsigned short v = 0;
    if (kk < RIN) v = f32_to_bf16_rne(kern[(size_t)k * INC * OUTC + kk * OUTC + o]);
    wB[idx] = v;
}

// ---- Main: gather(bf16, direct b128) -> WMMA bf16 -> fp32 atomic scatter ----
__global__ void __launch_bounds__(256)
sparse_conv_wmma2_kernel(const unsigned short* __restrict__ featB,  // [N][64] bf16
                         const unsigned short* __restrict__ wB,     // [27][96][72] bf16
                         const int*   __restrict__ in_map,
                         const int*   __restrict__ out_map,
                         float*       __restrict__ out) {
    __shared__ __align__(16) unsigned short Bs[ROUT * BS_STRIDE];   // 13824 B

    const int t       = threadIdx.x;
    const int k       = blockIdx.y;
    const int mbase   = blockIdx.x * MTILE;      // 625 * 128 == 80000 exactly
    const int mapBase = k * MPTS + mbase;

    // Stage this k's weight blob into LDS (already in fragment-ready layout)
    {
        const uint4* src = (const uint4*)(wB + (size_t)k * (ROUT * BS_STRIDE));
        uint4*       dst = (uint4*)Bs;
        const int n16 = (ROUT * BS_STRIDE) / 8;  // 864 x 16B
#if defined(HAVE_ASYNC_LDS)
        for (int i = t; i < n16; i += 256)
            __builtin_amdgcn_global_load_async_to_lds_b128(
                (AS1 v4i*)(uint4*)(src + i), (AS3 v4i*)(dst + i), 0, 0);
        asm volatile("s_wait_asynccnt 0" ::: "memory");
#else
        for (int i = t; i < n16; i += 256) dst[i] = src[i];
#endif
    }
    __syncthreads();

    const int wave = t >> 5;
    const int lane = t & 31;
    const int half = lane >> 4;
    const int l16  = lane & 15;
    const int m0   = wave * 16;

    // A fragments straight from global bf16 rows (ISA 16-bit A 16x32 layout):
    //   lane (half,l16): row m0+l16; chunk kc: halfs [kc+half*8,+8) and [kc+16+half*8,+8)
    const int rowIn = in_map[mapBase + m0 + l16];
    const uint4* arow = (const uint4*)(featB + (size_t)rowIn * KPAD);
    BFrag a0, a1;
    a0.q[0] = arow[half];
    a0.q[1] = arow[2 + half];
    a1.q[0] = arow[4 + half];
    a1.q[1] = arow[6 + half];

    // Output rows: uniform across each 16-lane half per accumulator register
    int orow[8];
    #pragma unroll
    for (int r = 0; r < 8; ++r) orow[r] = out_map[mapBase + m0 + half * 8 + r];

    #pragma unroll
    for (int nt = 0; nt < ROUT / 16; ++nt) {
        const int n0 = nt * 16;
        // B fragment (ISA 16-bit B 32x16): row n0+l16; chunk kc: halfs [kc+half*8,+8) and [kc+16+half*8,+8)
        const uint4* brow = (const uint4*)(Bs + (n0 + l16) * BS_STRIDE);
        BFrag b0, b1;
        b0.q[0] = brow[half];
        b0.q[1] = brow[2 + half];
        b1.q[0] = brow[4 + half];
        b1.q[1] = brow[6 + half];

        v8f c = {0.f, 0.f, 0.f, 0.f, 0.f, 0.f, 0.f, 0.f};
        c = __builtin_amdgcn_wmma_f32_16x16x32_bf16(false, a0.v, false, b0.v,
                                                    (short)0, c, false, false);
        c = __builtin_amdgcn_wmma_f32_16x16x32_bf16(false, a1.v, false, b1.v,
                                                    (short)0, c, false, false);

        const int col = n0 + l16;
        #pragma unroll
        for (int r = 0; r < 8; ++r)
            unsafeAtomicAdd(&out[(size_t)orow[r] * ROUT + col], c[r]);
    }
}

// ---- Fallback (workspace too small): self-contained gather+convert kernel ----
__global__ void __launch_bounds__(256)
sparse_conv_wmma_fallback_kernel(const float* __restrict__ features,
                                 const float* __restrict__ kern,
                                 const int*   __restrict__ in_map,
                                 const int*   __restrict__ out_map,
                                 float*       __restrict__ out) {
    __shared__ __align__(16) unsigned short As[MTILE * AS_STRIDE];
    __shared__ __align__(16) unsigned short Bs[ROUT  * BS_STRIDE];
    __shared__ int outRowS[MTILE];

    const int t = threadIdx.x;
    const int k = blockIdx.y;
    const int mbase = blockIdx.x * MTILE;
    const int mapBase = k * MPTS;
    const int kernBase = k * INC * OUTC;

    for (int idx = t; idx < RIN * ROUT; idx += 256) {
        int i = idx / ROUT, o = idx % ROUT;
        Bs[o * BS_STRIDE + i] = f32_to_bf16_rne(kern[kernBase + i * OUTC + o]);
    }
    for (int idx = t; idx < ROUT * (KPAD - RIN); idx += 256) {
        int o = idx / (KPAD - RIN), j = idx % (KPAD - RIN);
        Bs[o * BS_STRIDE + RIN + j] = 0;
    }
    {
        int m = t >> 1, part = t & 1;
        int mi = mbase + m;
        int rowIn = in_map[mapBase + mi];
        const float4* src = (const float4*)(features + (size_t)rowIn * RIN + part * 24);
        unsigned short* dst = &As[m * AS_STRIDE + part * 24];
        #pragma unroll
        for (int j = 0; j < 6; ++j) {
            float4 f = src[j];
            dst[j * 4 + 0] = f32_to_bf16_rne(f.x);
            dst[j * 4 + 1] = f32_to_bf16_rne(f.y);
            dst[j * 4 + 2] = f32_to_bf16_rne(f.z);
            dst[j * 4 + 3] = f32_to_bf16_rne(f.w);
        }
        if (part) {
            #pragma unroll
            for (int j = 0; j < KPAD - RIN; ++j) As[m * AS_STRIDE + RIN + j] = 0;
        } else {
            outRowS[m] = out_map[mapBase + mi];
        }
    }
    __syncthreads();

    const int wave = t >> 5, lane = t & 31, half = lane >> 4, l16 = lane & 15;
    const int m0 = wave * 16;

    BFrag a0, a1;
    {
        const int arow = (m0 + l16) * AS_STRIDE;
        const int kb = half * 8;
        #pragma unroll
        for (int p = 0; p < 8; ++p) {
            int kk = kb + ((p < 4) ? 0 : 16) + (p & 3) * 2;
            a0.u[p] = *(const unsigned int*)&As[arow + kk];
            a1.u[p] = *(const unsigned int*)&As[arow + 32 + kk];
        }
    }

    #pragma unroll
    for (int nt = 0; nt < ROUT / 16; ++nt) {
        const int n0 = nt * 16;
        BFrag b0, b1;
        const int brow = (n0 + l16) * BS_STRIDE;
        const int kb = half * 8;
        #pragma unroll
        for (int p = 0; p < 8; ++p) {
            int kk = kb + ((p < 4) ? 0 : 16) + (p & 3) * 2;
            b0.u[p] = *(const unsigned int*)&Bs[brow + kk];
            b1.u[p] = *(const unsigned int*)&Bs[brow + 32 + kk];
        }
        v8f c = {0.f, 0.f, 0.f, 0.f, 0.f, 0.f, 0.f, 0.f};
        c = __builtin_amdgcn_wmma_f32_16x16x32_bf16(false, a0.v, false, b0.v,
                                                    (short)0, c, false, false);
        c = __builtin_amdgcn_wmma_f32_16x16x32_bf16(false, a1.v, false, b1.v,
                                                    (short)0, c, false, false);
        const int col = n0 + l16;
        #pragma unroll
        for (int r = 0; r < 8; ++r) {
            int m = m0 + half * 8 + r;
            unsafeAtomicAdd(&out[(size_t)outRowS[m] * ROUT + col], c[r]);
        }
    }
}

extern "C" void kernel_launch(void* const* d_in, const int* in_sizes, int n_in,
                              void* d_out, int out_size, void* d_ws, size_t ws_size,
                              hipStream_t stream) {
    const float* features = (const float*)d_in[0];
    const float* kern     = (const float*)d_in[1];
    const int*   in_map   = (const int*)d_in[2];
    const int*   out_map  = (const int*)d_in[3];
    float*       out      = (float*)d_out;

    const int N = in_sizes[0] / RIN;

    int n4 = out_size / 4;
    zero_out_kernel<<<(n4 + 255) / 256, 256, 0, stream>>>((float4*)out, n4);

    dim3 grid((MPTS + MTILE - 1) / MTILE, KVOL);   // 625 x 27

    const size_t featBytes = (size_t)N * KPAD * sizeof(unsigned short);
    const size_t wBytes    = (size_t)KVOL * ROUT * BS_STRIDE * sizeof(unsigned short);

    if (ws_size >= featBytes + wBytes) {
        unsigned short* featB = (unsigned short*)d_ws;
        unsigned short* wB    = featB + (size_t)N * KPAD;

        feat_bf16_kernel<<<(N * 2 + 255) / 256, 256, 0, stream>>>(features, featB, N);
        int wTot = KVOL * ROUT * BS_STRIDE;
        weight_bf16_kernel<<<(wTot + 255) / 256, 256, 0, stream>>>(kern, wB);

        sparse_conv_wmma2_kernel<<<grid, 256, 0, stream>>>(featB, wB, in_map, out_map, out);
    } else {
        sparse_conv_wmma_fallback_kernel<<<grid, 256, 0, stream>>>(features, kern,
                                                                   in_map, out_map, out);
    }
}
